// InvariantPointAttention_36404142800962
// MI455X (gfx1250) — matrix operand order repaired and checked by reading, not compiled, not run on hardware
//
#include <hip/hip_runtime.h>
#include <hip/hip_bf16.h>
#include <math.h>

typedef __attribute__((ext_vector_type(16))) _Float16 v16h;
typedef __attribute__((ext_vector_type(8)))  float    v8f;

#define HEADS   12
#define HDIM    16
#define NQP     4
#define NVP     8
#define NODED   384
#define EDGED   128
#define NN      768
#define FIN_IN  2112   // 12*128 + 12*16 + 12*8*3 + 12*8

// ---- WMMA f16 fragment index helpers (CDNA5 ISA 7.12.2 layouts) ----
// A-matrix 16x32 f16: lane<16 -> K in {0..7,16..23}; lane>=16 -> K in {8..15,24..31}
__device__ __forceinline__ int a16_row(int lane) { return lane & 15; }
__device__ __forceinline__ int a16_col(int lane, int e) {
    return ((lane < 16) ? 0 : 8) + ((e >> 3) << 4) + (e & 7);
}
// B-matrix 32x16 f16: N = lane%16; lane<16 -> K in {0..7,16..23}; lane>=16 -> K in {8..15,24..31}
__device__ __forceinline__ int b16_row(int lane, int e) { return ((lane < 16) ? 0 : 8) + ((e >> 3) << 4) + (e & 7); }
__device__ __forceinline__ int b16_col(int lane) { return lane & 15; }
// C/D 16x16 f32: M = r + 8*(lane>=16), N = lane%16

// A fragment from a row-major f16 LDS tile: two contiguous 16B runs -> 2x ds_load_b128
__device__ __forceinline__ v16h load_a_lds(const _Float16* rowbase, int lane) {
    const int c0 = (lane < 16) ? 0 : 8;
    union { v16h v; uint4 q[2]; } u;
    u.q[0] = *(const uint4*)(rowbase + c0);        // elements 0..7  -> cols c0..c0+7
    u.q[1] = *(const uint4*)(rowbase + c0 + 16);   // elements 8..15 -> cols c0+16..c0+23
    return u.v;
}

// ---- gfx1250 async global->LDS (ASYNCcnt path) ----
#define ASYNC_LD_B128(ldsoff, gaddr) \
    asm volatile("global_load_async_to_lds_b128 %0, %1, off" :: "v"(ldsoff), "v"(gaddr) : "memory")
#define WAIT_ASYNC(n) asm volatile("s_wait_asynccnt %0" :: "i"(n) : "memory")

// =====================================================================
// K1: Y = node(768x384) @ {Wqkv(384x576), Wvqk(384x288), Wvv(384x288)}
// grid (48 row-tiles, 72 col-tiles), block = 1 wave (32 threads)
// =====================================================================
__global__ void k1_qkv(const float* __restrict__ node,
                       const float* __restrict__ Wqkv,
                       const float* __restrict__ Wvqk,
                       const float* __restrict__ Wvv,
                       float* __restrict__ qkvY,
                       float* __restrict__ vqkY,
                       float* __restrict__ vvY) {
    const int mt = blockIdx.x;          // 0..47
    const int ct = blockIdx.y;          // 0..71
    const int lane = threadIdx.x;       // wave32
    const float* W; float* Y; int ldw; int col0;
    if (ct < 36)      { W = Wqkv; Y = qkvY; ldw = 576; col0 = ct * 16; }
    else if (ct < 54) { W = Wvqk; Y = vqkY; ldw = 288; col0 = (ct - 36) * 16; }
    else              { W = Wvv;  Y = vvY;  ldw = 288; col0 = (ct - 54) * 16; }

    v8f c = {};
    const int row0 = mt * 16;
    for (int k0 = 0; k0 < NODED; k0 += 32) {
        v16h a, b;
        const int ar = row0 + a16_row(lane);
        #pragma unroll
        for (int e = 0; e < 16; ++e)
            a[e] = (_Float16)node[(size_t)ar * NODED + k0 + a16_col(lane, e)];
        const int bn = col0 + b16_col(lane);
        #pragma unroll
        for (int e = 0; e < 16; ++e)
            b[e] = (_Float16)W[(size_t)(k0 + b16_row(lane, e)) * ldw + bn];
        c = __builtin_amdgcn_wmma_f32_16x16x32_f16(false, a, false, b, (short)0, c, false, false);
    }
    const int n  = col0 + (lane & 15);
    const int mb = (lane < 16) ? 0 : 8;
    #pragma unroll
    for (int r = 0; r < 8; ++r)
        Y[(size_t)(row0 + mb + r) * ldw + n] = c[r];
}

// =====================================================================
// K2: per-node frame application: gq/gk/gvv = R*v + t, plus q_sq/k_sq
// grid 768, block 128
// =====================================================================
__global__ void k2_points(const float* __restrict__ vqkY, const float* __restrict__ vvY,
                          const float* __restrict__ R, const float* __restrict__ t,
                          float* __restrict__ gq, float* __restrict__ gk, float* __restrict__ gvv,
                          float* __restrict__ q_sq, float* __restrict__ k_sq) {
    const int n = blockIdx.x;
    const int tid = threadIdx.x;
    float Rm[9], tm[3];
    #pragma unroll
    for (int e = 0; e < 9; ++e) Rm[e] = R[(size_t)n * 9 + e];
    tm[0] = t[n * 3]; tm[1] = t[n * 3 + 1]; tm[2] = t[n * 3 + 2];

    if (tid < 24) {
        const int h = tid % 12, s = tid / 12;   // s=0 -> gq/q_sq, s=1 -> gk/k_sq
        const float* v = vqkY + (size_t)n * 288 + s * 144 + h * 12;
        float* g = (s == 0 ? gq : gk) + (size_t)(h * NN + n) * 12;
        float acc = 0.f;
        #pragma unroll
        for (int p = 0; p < NQP; ++p) {
            const float x = v[p * 3], y = v[p * 3 + 1], z = v[p * 3 + 2];
            const float g0 = Rm[0] * x + Rm[1] * y + Rm[2] * z + tm[0];
            const float g1 = Rm[3] * x + Rm[4] * y + Rm[5] * z + tm[1];
            const float g2 = Rm[6] * x + Rm[7] * y + Rm[8] * z + tm[2];
            g[p * 3] = g0; g[p * 3 + 1] = g1; g[p * 3 + 2] = g2;
            acc += g0 * g0 + g1 * g1 + g2 * g2;
        }
        (s == 0 ? q_sq : k_sq)[h * NN + n] = acc;
    } else if (tid < 24 + 96) {
        const int u = tid - 24, h = u >> 3, p = u & 7;
        const float* v = vvY + (size_t)n * 288 + h * 24 + p * 3;
        const float x = v[0], y = v[1], z = v[2];
        float* g = gvv + (size_t)(h * NN + n) * 24 + p * 3;
        g[0] = Rm[0] * x + Rm[1] * y + Rm[2] * z + tm[0];
        g[1] = Rm[3] * x + Rm[4] * y + Rm[5] * z + tm[1];
        g[2] = Rm[6] * x + Rm[7] * y + Rm[8] * z + tm[2];
    }
}

// =====================================================================
// K3: fused per-query-row attention (one HBM pass over edge_features,
// async global->LDS double-buffered streaming)
// grid 768, block 256 (8 waves)
// =====================================================================
#define SM_EDGE   0                    // _Float16[768*128]        196608 B
#define SM_LOG    196608               // float[12*768]             36864 B
#define SM_ATTN   233472               // _Float16[16*768]          24576 B
#define SM_STAGE  258048               // float[2][8192]            65536 B
#define SM_MISC   323584               // red/avv/rrinv/tv           2240 B
#define K3_SMEM   325824

__global__ void k3_row(const float* __restrict__ edge, const float* __restrict__ Webias,
                       const float* __restrict__ qkvY,
                       const float* __restrict__ gq, const float* __restrict__ gk,
                       const float* __restrict__ gvv,
                       const float* __restrict__ q_sq, const float* __restrict__ k_sq,
                       const float* __restrict__ R, const float* __restrict__ t,
                       float* __restrict__ combined) {
    const int i    = blockIdx.x;
    const int tid  = threadIdx.x;
    const int lane = tid & 31;
    const int wave = tid >> 5;

    extern __shared__ char smem[];
    _Float16* edgeH  = (_Float16*)(smem + SM_EDGE);
    float*    logits = (float*)(smem + SM_LOG);      // [h][j]
    _Float16* attnH  = (_Float16*)(smem + SM_ATTN);  // [16][768] (rows>=12 zero)
    float*    stage  = (float*)(smem + SM_STAGE);    // 2 x 8192 f32 staging
    float*    red    = (float*)(smem + SM_MISC);     // 256
    float*    avv    = red + 256;                    // 288
    float*    rrinv  = avv + 288;                    // 9
    float*    tv     = rrinv + 9;                    // 3

    // ---- 1. stream edge row: async global->LDS (f32 stage), cvt to f16 ----
    // 12 chunks of 64 j-rows (32 KB); each wave async-copies its 4 KB slice
    // (8 x b128 per lane), double-buffered so chunk c+1 streams while c converts.
    const char* erowB = (const char*)(edge + (size_t)i * (NN * EDGED));
    {
        #pragma unroll
        for (int it = 0; it < 8; ++it) {
            const unsigned lo = SM_STAGE + wave * 4096 + it * 512 + lane * 16;
            const char* ga = erowB + wave * 4096 + it * 512 + lane * 16;
            ASYNC_LD_B128(lo, ga);
        }
        for (int c = 0; c < 12; ++c) {
            if (c + 1 < 12) {
                const int buf = (c + 1) & 1;
                #pragma unroll
                for (int it = 0; it < 8; ++it) {
                    const unsigned lo = SM_STAGE + buf * 32768 + wave * 4096 + it * 512 + lane * 16;
                    const char* ga = erowB + (size_t)(c + 1) * 32768 + wave * 4096 + it * 512 + lane * 16;
                    ASYNC_LD_B128(lo, ga);
                }
                WAIT_ASYNC(8);     // in-order: chunk c's 8 are complete
            } else {
                WAIT_ASYNC(0);
            }
            __syncthreads();       // all waves' slices of chunk c landed
            const float* sb = stage + (c & 1) * 8192;
            for (int idx = tid; idx < 8192; idx += 256)
                edgeH[c * 8192 + idx] = (_Float16)sb[idx];
            __syncthreads();       // stage buffer free before reuse
        }
    }

    // ---- 2. edge_bias via WMMA: logits[h][j] = sum_d edge[j,d]*Webias[d,h] ----
    for (int jt = wave; jt < NN / 16; jt += 8) {
        v8f c = {};
        for (int k0 = 0; k0 < EDGED; k0 += 32) {
            const v16h a = load_a_lds(edgeH + (jt * 16 + a16_row(lane)) * EDGED + k0, lane);
            v16h b;
            const int bn = b16_col(lane);
            #pragma unroll
            for (int e = 0; e < 16; ++e) {
                const int d = k0 + b16_row(lane, e);
                b[e] = (bn < HEADS) ? (_Float16)Webias[d * HEADS + bn] : (_Float16)0.0f;
            }
            c = __builtin_amdgcn_wmma_f32_16x16x32_f16(false, a, false, b, (short)0, c, false, false);
        }
        const int h = lane & 15;
        if (h < HEADS) {
            const int jb = jt * 16 + ((lane < 16) ? 0 : 8);
            #pragma unroll
            for (int r = 0; r < 8; ++r) logits[h * NN + jb + r] = c[r];
        }
    }
    __syncthreads();

    // ---- 3. add node_scalar and point-distance terms ----
    const float inv_sqrt3 = 0.5773502691896258f;
    const float wcc = 0.11785113019775793f;          // sqrt(2/9/NQP)/2
    for (int u = tid; u < HEADS * NN; u += 256) {
        const int h = u / NN, j = u % NN;
        const float* sqv = qkvY + (size_t)i * 576 + h * 16;         // q (s=0)
        const float* skv = qkvY + (size_t)j * 576 + 192 + h * 16;   // k (s=1)
        float ns = 0.f;
        #pragma unroll
        for (int d = 0; d < HDIM; ++d) ns += sqv[d] * skv[d];
        ns *= 0.25f;                                                // d^-0.5
        const float* gki = gk + (size_t)(h * NN + i) * 12;
        const float* gqj = gq + (size_t)(h * NN + j) * 12;
        float cr = 0.f;
        #pragma unroll
        for (int e = 0; e < 12; ++e) cr += gki[e] * gqj[e];
        const float dist = (k_sq[h * NN + i] + q_sq[h * NN + j] - 2.f * cr) * wcc;
        logits[u] = (logits[u] + ns - dist) * inv_sqrt3;
    }
    __syncthreads();

    // ---- 4. softmax per head over j ----
    for (int h = 0; h < HEADS; ++h) {
        float m = -1e30f;
        for (int j = tid; j < NN; j += 256) m = fmaxf(m, logits[h * NN + j]);
        red[tid] = m; __syncthreads();
        for (int s = 128; s > 0; s >>= 1) { if (tid < s) red[tid] = fmaxf(red[tid], red[tid + s]); __syncthreads(); }
        m = red[0]; __syncthreads();
        float sum = 0.f;
        for (int j = tid; j < NN; j += 256) { float e = __expf(logits[h * NN + j] - m); logits[h * NN + j] = e; sum += e; }
        red[tid] = sum; __syncthreads();
        for (int s = 128; s > 0; s >>= 1) { if (tid < s) red[tid] += red[tid + s]; __syncthreads(); }
        const float inv = 1.f / red[0]; __syncthreads();
        for (int j = tid; j < NN; j += 256) {
            const float a = logits[h * NN + j] * inv;
            logits[h * NN + j] = a;
            attnH[h * NN + j] = (_Float16)a;
        }
    }
    for (int u = tid; u < 4 * NN; u += 256) attnH[HEADS * NN + u] = (_Float16)0.0f;
    __syncthreads();

    // ---- 5. edge_out via WMMA: C(16x16) = attn(16 x 768) @ edge(768 x 128) ----
    float* crow = combined + (size_t)i * FIN_IN;
    {
        const int nt = wave;                         // 8 waves, 8 N-tiles of d
        v8f c = {};
        for (int k0 = 0; k0 < NN; k0 += 32) {
            const v16h a = load_a_lds(attnH + a16_row(lane) * NN + k0, lane);
            v16h b;
            const int d = nt * 16 + b16_col(lane);
            #pragma unroll
            for (int e = 0; e < 16; ++e) {
                const int j = k0 + b16_row(lane, e);
                b[e] = edgeH[j * EDGED + d];
            }
            c = __builtin_amdgcn_wmma_f32_16x16x32_f16(false, a, false, b, (short)0, c, false, false);
        }
        const int d  = nt * 16 + (lane & 15);
        const int hb = (lane < 16) ? 0 : 8;
        #pragma unroll
        for (int r = 0; r < 8; ++r) {
            const int h = hb + r;
            if (h < HEADS) crow[h * EDGED + d] = c[r];
        }
    }

    // ---- 6. scalar_out and att_v (small contractions, VALU) ----
    for (int u = tid; u < 480; u += 256) {
        if (u < 192) {                               // scalar_out: 12 heads x 16 dims
            const int h = u >> 4, d = u & 15;
            float s = 0.f;
            const float* svc = qkvY + 384 + h * 16 + d;   // s=2 offset within row
            for (int j = 0; j < NN; ++j) s += logits[h * NN + j] * svc[(size_t)j * 576];
            crow[1536 + h * 16 + d] = s;
        } else {                                     // att_v: 12 x 8 x 3
            const int v = u - 192;
            const int h = v / 24, rem = v % 24, p = rem / 3, cc = rem % 3;
            float s = 0.f;
            const float* g = gvv + (size_t)h * NN * 24 + p * 3 + cc;
            for (int j = 0; j < NN; ++j) s += logits[h * NN + j] * g[(size_t)j * 24];
            avv[v] = s;
        }
    }
    if (tid == 0) {                                  // 3x3 inverse via adjugate
        const float* Ri = R + (size_t)i * 9;
        const float a00=Ri[0],a01=Ri[1],a02=Ri[2],a10=Ri[3],a11=Ri[4],a12=Ri[5],a20=Ri[6],a21=Ri[7],a22=Ri[8];
        const float det = a00*(a11*a22-a12*a21) - a01*(a10*a22-a12*a20) + a02*(a10*a21-a11*a20);
        const float id = 1.f / det;
        rrinv[0]=(a11*a22-a12*a21)*id; rrinv[1]=(a02*a21-a01*a22)*id; rrinv[2]=(a01*a12-a02*a11)*id;
        rrinv[3]=(a12*a20-a10*a22)*id; rrinv[4]=(a00*a22-a02*a20)*id; rrinv[5]=(a02*a10-a00*a12)*id;
        rrinv[6]=(a10*a21-a11*a20)*id; rrinv[7]=(a01*a20-a00*a21)*id; rrinv[8]=(a00*a11-a01*a10)*id;
        tv[0]=t[i*3]; tv[1]=t[i*3+1]; tv[2]=t[i*3+2];
    }
    __syncthreads();

    // ---- 7. local frame + vector length ----
    if (tid < 96) {
        const int h = tid >> 3, p = tid & 7;
        const float x = avv[h*24+p*3+0] - tv[0];
        const float y = avv[h*24+p*3+1] - tv[1];
        const float z = avv[h*24+p*3+2] - tv[2];
        const float l0 = rrinv[0]*x + rrinv[1]*y + rrinv[2]*z;
        const float l1 = rrinv[3]*x + rrinv[4]*y + rrinv[5]*z;
        const float l2 = rrinv[6]*x + rrinv[7]*y + rrinv[8]*z;
        const int o = 1728 + (h * 8 + p) * 3;
        crow[o] = l0; crow[o+1] = l1; crow[o+2] = l2;
        crow[2016 + h * 8 + p] = sqrtf(l0*l0 + l1*l1 + l2*l2);
    }
}

// =====================================================================
// K4: out = node + combined(768x2112) @ Wfin(2112x384) + bfin
// grid (48, 24), block = 1 wave
// =====================================================================
__global__ void k4_final(const float* __restrict__ combined, const float* __restrict__ Wfin,
                         const float* __restrict__ bfin, const float* __restrict__ node,
                         float* __restrict__ out) {
    const int mt = blockIdx.x, nt = blockIdx.y;
    const int lane = threadIdx.x;
    const int row0 = mt * 16, col0 = nt * 16;
    v8f c = {};
    for (int k0 = 0; k0 < FIN_IN; k0 += 32) {
        v16h a, b;
        const int ar = row0 + a16_row(lane);
        #pragma unroll
        for (int e = 0; e < 16; ++e)
            a[e] = (_Float16)combined[(size_t)ar * FIN_IN + k0 + a16_col(lane, e)];
        const int bn = col0 + b16_col(lane);
        #pragma unroll
        for (int e = 0; e < 16; ++e)
            b[e] = (_Float16)Wfin[(size_t)(k0 + b16_row(lane, e)) * NODED + bn];
        c = __builtin_amdgcn_wmma_f32_16x16x32_f16(false, a, false, b, (short)0, c, false, false);
    }
    const int n  = col0 + (lane & 15);
    const int mb = (lane < 16) ? 0 : 8;
    #pragma unroll
    for (int r = 0; r < 8; ++r) {
        const int m = row0 + mb + r;
        out[(size_t)m * NODED + n] = node[(size_t)m * NODED + n] + c[r] + bfin[n];
    }
}

// =====================================================================
extern "C" void kernel_launch(void* const* d_in, const int* in_sizes, int n_in,
                              void* d_out, int out_size, void* d_ws, size_t ws_size,
                              hipStream_t stream) {
    const float* node   = (const float*)d_in[0];
    const float* edge   = (const float*)d_in[1];
    const float* R      = (const float*)d_in[2];
    const float* t      = (const float*)d_in[3];
    const float* Wqkv   = (const float*)d_in[4];
    const float* Wvqk   = (const float*)d_in[5];
    const float* Wvv    = (const float*)d_in[6];
    const float* Webias = (const float*)d_in[7];
    const float* Wfin   = (const float*)d_in[8];
    const float* bfin   = (const float*)d_in[9];
    float* out = (float*)d_out;

    // workspace layout (floats)
    float* ws   = (float*)d_ws;
    float* qkvY = ws;                      // 768*576   = 442368
    float* vqkY = qkvY + 442368;           // 768*288   = 221184
    float* vvY  = vqkY + 221184;           // 768*288   = 221184
    float* gq   = vvY  + 221184;           // 12*768*12 = 110592
    float* gk   = gq   + 110592;           // 110592
    float* gvv  = gk   + 110592;           // 12*768*24 = 221184
    float* q_sq = gvv  + 221184;           // 9216
    float* k_sq = q_sq + 9216;             // 9216
    float* combined = k_sq + 9216;         // 768*2112  = 1622016  (total ~11.9 MB)

    dim3 g1(48, 72);
    k1_qkv<<<g1, 32, 0, stream>>>(node, Wqkv, Wvqk, Wvv, qkvY, vqkY, vvY);
    k2_points<<<768, 128, 0, stream>>>(vqkY, vvY, R, t, gq, gk, gvv, q_sq, k_sq);
    k3_row<<<768, 256, K3_SMEM, stream>>>(edge, Webias, qkvY, gq, gk, gvv, q_sq, k_sq, R, t, combined);
    dim3 g4(48, 24);
    k4_final<<<g4, 32, 0, stream>>>(combined, Wfin, bfin, node, out);
}
